// CasualAttention_3143916060973
// MI455X (gfx1250) — matrix operand order repaired and checked by reading, not compiled, not run on hardware
//
#include <hip/hip_runtime.h>
#include <stdint.h>

// Causal attention for B=4, S=4096, D=512 on gfx1250 (MI455X).
// Compute-bound (~130 GFLOP vs ~100MB traffic) -> bf16 WMMA w/ fp32 accum.

#define B_  4
#define S_  4096
#define D_  512

typedef __attribute__((ext_vector_type(16))) __bf16 v16bf;
typedef __attribute__((ext_vector_type(8)))  float  v8f;

union FragU {
  v16bf   v;
  uint4   q[2];
  uint32_t u[8];
};

__device__ __forceinline__ uint32_t f2bf(float f) {
  uint32_t u = __float_as_uint(f);
  return (u + 0x7FFFu + ((u >> 16) & 1u)) >> 16;   // round-to-nearest-even
}

// A-matrix 16x32 bf16 fragment (row-major source; lane holds row lane&15).
// ISA layout: lanes 0-15: V0..3 = K0..7 pairs, V4..7 = K16..23;
//             lanes16-31: V0..3 = K8..15,      V4..7 = K24..31.
__device__ __forceinline__ v16bf load_fragA(const uint16_t* base, int ld, int lane) {
  const int row = lane & 15, hb = lane >> 4;
  const uint16_t* p = base + row * ld + hb * 8;
  FragU f;
  f.q[0] = *(const uint4*)(p);        // K = hb*8 .. hb*8+7
  f.q[1] = *(const uint4*)(p + 16);   // K = 16+hb*8 .. +7
  return f.v;
}

// B-matrix 32x16 bf16 fragment. Source row n holds contiguous k.
// lanes 0-15 hold K=0..7,16..23 (2/VGPR), lanes 16-31 hold K=8..15,24..31.
__device__ __forceinline__ v16bf load_fragB(const uint16_t* base, int ld, int lane) {
  const int col = lane & 15, hb = lane >> 4;
  const uint16_t* p = base + col * ld + hb * 8;
  FragU f;
  f.q[0] = *(const uint4*)(p);        // K = hb*8 .. hb*8+7
  f.q[1] = *(const uint4*)(p + 16);   // K = 16+hb*8 .. +7
  return f.v;
}

__device__ __forceinline__ v8f wmma_bf16(v16bf a, v16bf b, v8f c) {
  return __builtin_amdgcn_wmma_f32_16x16x32_bf16(false, a, false, b, (short)0, c,
                                                 false, false);
}

// ---------------- fp32 -> bf16 conversion ----------------
__global__ void cvt_bf16_kernel(const float* __restrict__ src,
                                uint16_t* __restrict__ dst, int n4) {
  int i = blockIdx.x * blockDim.x + threadIdx.x;
  if (i < n4) {
    float4 f = ((const float4*)src)[i];
    uint32_t a = f2bf(f.x) | (f2bf(f.y) << 16);
    uint32_t b = f2bf(f.z) | (f2bf(f.w) << 16);
    ((uint2*)dst)[i] = make_uint2(a, b);
  }
}

// ---------------- QKV projection: y = x @ W^T -------------
// z=0 -> Q [BS, D], z=1 -> K [BS, D], z=2 -> V^T [B, D, S]
__global__ __launch_bounds__(256) void qkv_gemm_kernel(
    const uint16_t* __restrict__ x, const uint16_t* __restrict__ w3,
    uint16_t* __restrict__ Qo, uint16_t* __restrict__ Ko,
    uint16_t* __restrict__ Vt) {
  const int z    = blockIdx.z;
  const uint16_t* w = w3 + z * (D_ * D_);
  const int tid  = threadIdx.x;
  const int lane = tid & 31, wave = tid >> 5;
  const int m0 = blockIdx.x * 64 + (wave >> 2) * 32;   // rows of x
  const int n0 = blockIdx.y * 128 + (wave & 3) * 32;   // rows of W (out cols)

  v8f cc[2][2];
#pragma unroll
  for (int a = 0; a < 2; a++)
#pragma unroll
    for (int b = 0; b < 2; b++) cc[a][b] = (v8f){0, 0, 0, 0, 0, 0, 0, 0};

  for (int k0 = 0; k0 < D_; k0 += 32) {
    v16bf a0 = load_fragA(x + (size_t)m0 * D_ + k0, D_, lane);
    v16bf a1 = load_fragA(x + (size_t)(m0 + 16) * D_ + k0, D_, lane);
    v16bf b0 = load_fragB(w + (size_t)n0 * D_ + k0, D_, lane);
    v16bf b1 = load_fragB(w + (size_t)(n0 + 16) * D_ + k0, D_, lane);
    cc[0][0] = wmma_bf16(a0, b0, cc[0][0]);
    cc[0][1] = wmma_bf16(a0, b1, cc[0][1]);
    cc[1][0] = wmma_bf16(a1, b0, cc[1][0]);
    cc[1][1] = wmma_bf16(a1, b1, cc[1][1]);
  }

  uint16_t* outp = (z == 0) ? Qo : (z == 1) ? Ko : Vt;
  const int hb = lane >> 4, col = lane & 15;
#pragma unroll
  for (int tm = 0; tm < 2; tm++)
#pragma unroll
    for (int tn = 0; tn < 2; tn++)
#pragma unroll
      for (int r = 0; r < 8; r++) {
        int m = m0 + tm * 16 + r + 8 * hb;
        int n = n0 + tn * 16 + col;
        uint16_t val = (uint16_t)f2bf(cc[tm][tn][r]);
        if (z < 2) {
          outp[(size_t)m * D_ + n] = val;                       // [BS, D]
        } else {
          int batch = m >> 12, s = m & (S_ - 1);
          outp[(size_t)batch * D_ * S_ + (size_t)n * S_ + s] = val;  // [B, D, S]
        }
      }
}

// ---------------- flash attention ----------------
// Block: 256 threads = 2 quads. Quad g: 16 queries; wave wq within quad owns
// output D-slice [wq*128, +128). Scores computed as partial-d and summed via LDS.
__global__ __launch_bounds__(256) void attn_kernel(
    const uint16_t* __restrict__ Q, const uint16_t* __restrict__ K,
    const uint16_t* __restrict__ VT, float* __restrict__ out) {
  __shared__ float pex[2][4][16][66];   // [quad][wave-in-quad][row][key] partial scores

  const int b   = blockIdx.y;
  const int q0  = blockIdx.x * 32;
  const int tid = threadIdx.x;
  const int lane = tid & 31, wave = tid >> 5;
  const int g = wave >> 2, wq = wave & 3;
  const int qbase = q0 + g * 16;
  const int dbase = wq * 128;
  const int row = lane & 15, hb = lane >> 4;

  const uint16_t* Qb = Q + (size_t)b * S_ * D_;
  const uint16_t* Kb = K + (size_t)b * S_ * D_;
  const uint16_t* Vb = VT + (size_t)b * D_ * S_;

  // resident Q fragments for this wave's d-slice (k = d)
  v16bf qf[4];
#pragma unroll
  for (int ks = 0; ks < 4; ks++)
    qf[ks] = load_fragA(Qb + (size_t)qbase * D_ + dbase + ks * 32, D_, lane);

  v8f acc[8];
#pragma unroll
  for (int t = 0; t < 8; t++) acc[t] = (v8f){0, 0, 0, 0, 0, 0, 0, 0};

  float m_run = -1e30f, l_run = 0.f;
  const float SCL = 0.06376353717982492f;   // log2(e) / sqrt(512)
  const int nkt = (q0 >> 6) + 1;            // causal key-tile count (64 keys/tile)

  for (int kt = 0; kt < nkt; kt++) {
    const int kbase = kt * 64;

    // 1) partial scores over this wave's d-slice: S_part = Q_slice @ K_slice^T
#pragma unroll
    for (int nt = 0; nt < 4; nt++) {
      v8f sc = (v8f){0, 0, 0, 0, 0, 0, 0, 0};
#pragma unroll
      for (int ks = 0; ks < 4; ks++) {
        v16bf bf = load_fragB(Kb + (size_t)(kbase + nt * 16) * D_ + dbase + ks * 32,
                              D_, lane);
        sc = wmma_bf16(qf[ks], bf, sc);
      }
#pragma unroll
      for (int r = 0; r < 8; r++)
        pex[g][wq][r + 8 * hb][nt * 16 + row] = sc[r];
    }
    __syncthreads();

    // 2) gather full scores (sum 4 partials), causal mask, lane-local softmax.
    //    Lane holds its A-fragment positions of row `row`.
    float pv[32];
    float mt = -1e30f;
#pragma unroll
    for (int f = 0; f < 2; f++)
#pragma unroll
      for (int i = 0; i < 8; i++) {
        const int kl = f * 32 + ((i & 4) ? 16 : 0) + (hb << 3) + ((i & 3) << 1);
        float2 p0 = *(const float2*)&pex[g][0][row][kl];
        float2 p1 = *(const float2*)&pex[g][1][row][kl];
        float2 p2 = *(const float2*)&pex[g][2][row][kl];
        float2 p3 = *(const float2*)&pex[g][3][row][kl];
        float s0 = (p0.x + p1.x + p2.x + p3.x) * SCL;
        float s1 = (p0.y + p1.y + p2.y + p3.y) * SCL;
        const int kg = kbase + kl, qg = qbase + row;
        if (kg > qg) s0 = -1e30f;
        if (kg + 1 > qg) s1 = -1e30f;
        pv[f * 16 + 2 * i] = s0;
        pv[f * 16 + 2 * i + 1] = s1;
        mt = fmaxf(mt, fmaxf(s0, s1));
      }
    mt = fmaxf(mt, __shfl_xor(mt, 16, 32));        // row split across lanes L, L+16
    const float m_new = fmaxf(m_run, mt);
    const float alpha = exp2f(m_run - m_new);
    m_run = m_new;

    float lsum = 0.f;
    FragU pf[2];
#pragma unroll
    for (int f = 0; f < 2; f++)
#pragma unroll
      for (int i = 0; i < 8; i++) {
        float e0 = exp2f(pv[f * 16 + 2 * i] - m_new);
        float e1 = exp2f(pv[f * 16 + 2 * i + 1] - m_new);
        lsum += e0 + e1;
        pf[f].u[i] = f2bf(e0) | (f2bf(e1) << 16);  // bf16 A-fragment of P
      }
    l_run = l_run * alpha + lsum;

    // 3) rescale running output by alpha (broadcast per accumulator row)
#pragma unroll
    for (int r = 0; r < 8; r++) {
      float ar = __shfl(alpha, r + 8 * hb, 32);
#pragma unroll
      for (int t = 0; t < 8; t++) acc[t][r] *= ar;
    }

    // 4) PV: acc(16 x 128) += P(16 x 64) @ V(64 x 128-slice), V^T rows contiguous in k
#pragma unroll
    for (int t = 0; t < 8; t++)
#pragma unroll
      for (int f = 0; f < 2; f++) {
        v16bf vf = load_fragB(Vb + (size_t)(dbase + t * 16) * S_ + kbase + f * 32,
                              S_, lane);
        acc[t] = wmma_bf16(pf[f].v, vf, acc[t]);
      }
    __syncthreads();
  }

  // finalize: divide by row sums and store fp32
  float lt = l_run + __shfl_xor(l_run, 16, 32);
  float rl = 1.0f / lt;
  float* ob = out + (size_t)b * S_ * D_;
#pragma unroll
  for (int r = 0; r < 8; r++) {
    float rr = __shfl(rl, r + 8 * hb, 32);
    int m = qbase + r + 8 * hb;
#pragma unroll
    for (int t = 0; t < 8; t++)
      ob[(size_t)m * D_ + dbase + t * 16 + row] = acc[t][r] * rr;
  }
}

// ---------------- launcher ----------------
extern "C" void kernel_launch(void* const* d_in, const int* in_sizes, int n_in,
                              void* d_out, int out_size, void* d_ws, size_t ws_size,
                              hipStream_t stream) {
  (void)in_sizes; (void)n_in; (void)out_size; (void)ws_size;
  const float* x  = (const float*)d_in[0];
  const float* Wq = (const float*)d_in[1];
  const float* Wk = (const float*)d_in[2];
  const float* Wv = (const float*)d_in[3];
  float* out = (float*)d_out;

  char* ws = (char*)d_ws;
  const size_t XB = (size_t)B_ * S_ * D_ * 2;       // 16 MB
  const size_t WB = (size_t)D_ * D_ * 2;            // 0.5 MB each
  uint16_t* xb = (uint16_t*)ws;
  uint16_t* wb = (uint16_t*)(ws + XB);
  uint16_t* Qb = (uint16_t*)(ws + XB + 3 * WB);
  uint16_t* Kb = (uint16_t*)(ws + XB + 3 * WB + XB);
  uint16_t* Vt = (uint16_t*)(ws + XB + 3 * WB + 2 * XB);

  const int n4x = (B_ * S_ * D_) / 4;
  cvt_bf16_kernel<<<(n4x + 255) / 256, 256, 0, stream>>>(x, xb, n4x);
  const int n4w = (D_ * D_) / 4;
  cvt_bf16_kernel<<<(n4w + 255) / 256, 256, 0, stream>>>(Wq, wb, n4w);
  cvt_bf16_kernel<<<(n4w + 255) / 256, 256, 0, stream>>>(Wk, wb + D_ * D_, n4w);
  cvt_bf16_kernel<<<(n4w + 255) / 256, 256, 0, stream>>>(Wv, wb + 2 * D_ * D_, n4w);

  // Q/K/V projections: grid (M/64, N/128, 3)
  qkv_gemm_kernel<<<dim3((B_ * S_) / 64, D_ / 128, 3), 256, 0, stream>>>(
      xb, wb, Qb, Kb, Vt);

  // flash attention: grid (S/32, B)
  attn_kernel<<<dim3(S_ / 32, B_), 256, 0, stream>>>(Qb, Kb, Vt, out);
}
